// Attention_23699629539334
// MI455X (gfx1250) — matrix operand order repaired and checked by reading, not compiled, not run on hardware
//
#include <hip/hip_runtime.h>
#include <hip/hip_bf16.h>
#include <cstdint>

typedef unsigned short u16;
typedef unsigned int   u32;

typedef __attribute__((ext_vector_type(16))) __bf16 v16bf;
typedef __attribute__((ext_vector_type(8)))  float  v8f;
typedef __attribute__((ext_vector_type(4)))  u32    u32x4;

union BF16x16 {
  v16bf v;
  u16   u[16];
  u32x4 q[2];
};

constexpr int   CB = 2, CS = 1024, CE = 2048, CH = 16, CKVH = 4, CD = 128;
constexpr int   NQ = 4096, NKC = 2048, NVC = 512, NTOT = 6656;
constexpr float NEGF = -1e30f;
constexpr float INV_SQRT_D = 0.08838834764831845f;  // 1/sqrt(128)
constexpr float LAM_INIT = 0.2f;

__device__ __forceinline__ u16 f32_to_bf16(float f) {
  union { float f; u32 u; } x; x.f = f;
  u32 r = x.u + 0x7FFFu + ((x.u >> 16) & 1u);   // round-to-nearest-even
  return (u16)(r >> 16);
}
__device__ __forceinline__ float bf16_to_f32(u16 h) {
  union { u32 u; float f; } x; x.u = ((u32)h) << 16;
  return x.f;
}
__device__ __forceinline__ float wave_max(float v) {
  #pragma unroll
  for (int m = 16; m > 0; m >>= 1) v = fmaxf(v, __shfl_xor(v, m, 32));
  return v;
}
__device__ __forceinline__ float wave_sum(float v) {
  #pragma unroll
  for (int m = 16; m > 0; m >>= 1) v += __shfl_xor(v, m, 32);
  return v;
}
__device__ __forceinline__ v8f wmma_bf16(const BF16x16& a, const BF16x16& b, v8f c) {
  return __builtin_amdgcn_wmma_f32_16x16x32_bf16(false, a.v, false, b.v,
                                                 (short)0, c, false, false);
}

// ---------------------------------------------------------------------------
// Pre-pass 1: x (fp32) -> xbf (bf16), straight streaming conversion.
// ---------------------------------------------------------------------------
__global__ void cvt_x_bf16(const float* __restrict__ x, u16* __restrict__ xbf) {
  int i = blockIdx.x * blockDim.x + threadIdx.x;     // pair index
  float2 f = ((const float2*)x)[i];
  u32 lo = (u32)f32_to_bf16(f.x);
  u32 hi = (u32)f32_to_bf16(f.y);
  ((u32*)xbf)[i] = lo | (hi << 16);
}

// ---------------------------------------------------------------------------
// Pre-pass 2: W (K=2048 x N fp32, row-major) -> WTdst (N x 2048 bf16, n-major)
// 32x32 tile transpose through padded LDS; both phases fully coalesced.
// ---------------------------------------------------------------------------
__global__ void transpose_w_bf16(const float* __restrict__ W, int N,
                                 u16* __restrict__ WTdst) {
  __shared__ u16 sT[32][33];
  const int kk = blockIdx.x * 32, nn = blockIdx.y * 32;
  const int tid = threadIdx.x;
  #pragma unroll
  for (int p = 0; p < 4; ++p) {
    int flat = tid + p * 256;
    int lk = flat >> 5, ln = flat & 31;
    sT[ln][lk] = f32_to_bf16(W[(size_t)(kk + lk) * N + nn + ln]);
  }
  __syncthreads();
  #pragma unroll
  for (int p = 0; p < 4; ++p) {
    int flat = tid + p * 256;
    int ln = flat >> 5, lk = flat & 31;
    WTdst[(size_t)(nn + ln) * CE + kk + lk] = sT[ln][lk];
  }
}

// ---------------------------------------------------------------------------
// Epilogue scatter for the QKV GEMM (one 16x16 tile in ISA C/D layout).
// ---------------------------------------------------------------------------
__device__ __forceinline__ void store_qkv_tile(int m0, int lane, int ng, const v8f& c,
                                               u16* __restrict__ q_raw,
                                               u16* __restrict__ k_raw,
                                               u16* __restrict__ vT) {
  #pragma unroll
  for (int j = 0; j < 8; ++j) {
    int m = m0 + ((lane < 16) ? j : j + 8);
    u16 hv = f32_to_bf16(c[j]);
    if (ng < NQ) {
      int h = ng >> 8, d = ng & 255;
      q_raw[(size_t)(m * CH + h) * 256 + d] = hv;
    } else if (ng < NQ + NKC) {
      int nk = ng - NQ; int h = nk >> 8, d = nk & 255;
      k_raw[(size_t)(m * CKVH + h) * 256 + d] = hv;
    } else {
      int nv = ng - NQ - NKC; int kvh = nv >> 7, d = nv & 127;
      int bb = m >> 10, ss = m & 1023;
      vT[(((size_t)bb * CKVH + kvh) * CD + d) * CS + ss] = hv;  // V transposed
    }
  }
}

// ---------------------------------------------------------------------------
// Kernel: fused QKV GEMM. C(2048 x 6656) = xbf(2048 x 2048) * WT^T.
// All-bf16 operands, no LDS, no conversions in the loop: each wave owns a
// 16x32 output slab (2 accumulators), A fragment reused across 2 WMMAs.
// grid (128, 52), block 128 (4 waves -> 128 cols per block).
// ---------------------------------------------------------------------------
__global__ void qkv_gemm(const u16* __restrict__ xbf,
                         const u16* __restrict__ WT,
                         u16* __restrict__ q_raw,
                         u16* __restrict__ k_raw,
                         u16* __restrict__ vT) {
  const int tid = threadIdx.x, wave = tid >> 5, lane = tid & 31;
  const int m0 = blockIdx.x * 16;
  const int nw = blockIdx.y * 128 + wave * 32;
  const int off1 = (lane < 16) ? 0 : 16;       // contiguous k order (matches B)
  const int bo16 = (lane < 16) ? 0 : 16;       // contiguous k order (matches A)
  const u16* ap = xbf + (size_t)(m0 + (lane & 15)) * CE;
  const u16* b0 = WT + (size_t)(nw + (lane & 15)) * CE;
  const u16* b1 = WT + (size_t)(nw + 16 + (lane & 15)) * CE;

  v8f c0 = {}, c1 = {};
  for (int k0 = 0; k0 < CE; k0 += 32) {
    if (k0 + 64 < CE) {
      // locality 3 -> WGP-scope prefetch: pull B rows into all cache levels
      __builtin_prefetch(b0 + k0 + 64, 0, 3);
      __builtin_prefetch(b1 + k0 + 64, 0, 3);
    }
    BF16x16 a, f0, f1;
    a.q[0]  = *(const u32x4*)(ap + k0 + off1);
    a.q[1]  = *(const u32x4*)(ap + k0 + off1 + 8);
    f0.q[0] = *(const u32x4*)(b0 + k0 + bo16);
    f0.q[1] = *(const u32x4*)(b0 + k0 + bo16 + 8);
    f1.q[0] = *(const u32x4*)(b1 + k0 + bo16);
    f1.q[1] = *(const u32x4*)(b1 + k0 + bo16 + 8);
    c0 = wmma_bf16(a, f0, c0);
    c1 = wmma_bf16(a, f1, c1);
  }
  store_qkv_tile(m0, lane, nw + (lane & 15),      c0, q_raw, k_raw, vT);
  store_qkv_tile(m0, lane, nw + 16 + (lane & 15), c1, q_raw, k_raw, vT);
}

// ---------------------------------------------------------------------------
// Kernel: RoPE on q_raw and k_raw in place.
// ---------------------------------------------------------------------------
__global__ void rope_kernel(u16* __restrict__ q_raw, u16* __restrict__ k_raw) {
  constexpr int QP = CB * CS * CH * CD;    // 4194304 q pairs
  constexpr int KP = CB * CS * CKVH * CD;  // 1048576 k pairs
  int idx = blockIdx.x * blockDim.x + threadIdx.x;
  if (idx >= QP + KP) return;
  u16* buf; size_t base; int j, s;
  if (idx < QP) {
    int p = idx; j = p & 127; p >>= 7;
    int h = p & 15; p >>= 4;
    s = p & 1023; int b = p >> 10;
    base = ((size_t)((((b << 10) | s) * CH + h))) * 256 + 2 * j;
    buf = q_raw;
  } else {
    int p = idx - QP; j = p & 127; p >>= 7;
    int h = p & 3; p >>= 2;
    s = p & 1023; int b = p >> 10;
    base = ((size_t)((((b << 10) | s) * CKVH + h))) * 256 + 2 * j;
    buf = k_raw;
  }
  float theta = __expf(-(float)j * (9.210340371976184f / 128.0f)); // ln(1e4)/128
  float ang = (float)s * theta;
  float sn, cs;
  __sincosf(ang, &sn, &cs);
  float re = bf16_to_f32(buf[base]);
  float im = bf16_to_f32(buf[base + 1]);
  buf[base]     = f32_to_bf16(re * cs - im * sn);
  buf[base + 1] = f32_to_bf16(re * sn + im * cs);
}

// ---------------------------------------------------------------------------
// Kernel: per-head lambda = exp(lq1.lk1) - exp(lq2.lk2) + LAM_INIT
// ---------------------------------------------------------------------------
__global__ void lambda_kernel(const float* __restrict__ lq1,
                              const float* __restrict__ lq2,
                              const float* __restrict__ lk1,
                              const float* __restrict__ lk2,
                              float* __restrict__ lam) {
  int h = threadIdx.x;
  if (h < CH) {
    float d1 = 0.f, d2 = 0.f;
    for (int i = 0; i < CD; ++i) {
      d1 += lq1[h * CD + i] * lk1[h * CD + i];
      d2 += lq2[h * CD + i] * lk2[h * CD + i];
    }
    lam[h] = __expf(d1) - __expf(d2) + LAM_INIT;
  }
}

// ---------------------------------------------------------------------------
// Kernel: differential attention. One block (8 waves) per (b, h, 16-query
// tile). Full 2x16x1024 energy kept in LDS (168 KB -> CDNA5 320KB WGP).
// ---------------------------------------------------------------------------
__global__ void diff_attn_kernel(const u16* __restrict__ q_raw,
                                 const u16* __restrict__ k_raw,
                                 const u16* __restrict__ vT,
                                 const float* __restrict__ lam_ws,
                                 const float* __restrict__ rms_w,
                                 u16* __restrict__ att_ws) {
  extern __shared__ char smem[];
  float* sE    = (float*)smem;                                  // [32][1024] f32
  u16*   sDiff = (u16*)(smem + 32 * 1024 * 4);                  // [16][1024] bf16
  float* sOut  = (float*)(smem + 32 * 1024 * 4 + 16 * 1024 * 2);// [16][128] f32

  const int tid = threadIdx.x, wave = tid >> 5, lane = tid & 31;
  const int qblk = blockIdx.x, h = blockIdx.y, b = blockIdx.z;
  const int kvh = h >> 2;                      // GQA ratio 4
  const int q0 = qblk * 16;
  const float lam = lam_ws[h];
  const int off1 = (lane < 16) ? 0 : 16;
  const int bo16 = (lane < 16) ? 0 : 16;

  // ---- Phase 1: energy[a][q][k] = qd . kd, causal-masked, into LDS ----
  for (int aH = 0; aH < 2; ++aH) {
    const int srow = q0 + (lane & 15);
    const u16* qrow = q_raw + ((size_t)((((b << 10) + srow) * CH + h))) * 256 + aH * CD;
    BF16x16 afr[4];                            // K=128 in 4 steps of 32
    #pragma unroll
    for (int ks = 0; ks < 4; ++ks) {
      const u16* p = qrow + ks * 32 + off1;
      afr[ks].q[0] = *(const u32x4*)(p);
      afr[ks].q[1] = *(const u32x4*)(p + 8);
    }
    for (int kb = wave; kb < 64; kb += 8) {    // 8 key-blocks per wave
      const int key = kb * 16 + (lane & 15);
      const u16* krow = k_raw + ((size_t)((((b << 10) + key) * CKVH + kvh))) * 256 + aH * CD;
      v8f c = {};
      #pragma unroll
      for (int ks = 0; ks < 4; ++ks) {
        const u16* p = krow + ks * 32 + bo16;
        BF16x16 bfr;
        bfr.q[0] = *(const u32x4*)(p);
        bfr.q[1] = *(const u32x4*)(p + 8);
        c = wmma_bf16(afr[ks], bfr, c);
      }
      #pragma unroll
      for (int j = 0; j < 8; ++j) {
        int qr = (lane < 16) ? j : j + 8;
        float val = (key > q0 + qr) ? NEGF : c[j];
        sE[(aH * 16 + qr) * 1024 + key] = val;
      }
    }
  }
  __syncthreads();

  // ---- Phase 2: softmax(energy / sqrt(D)) for all 32 (half,q) rows ----
  #pragma unroll
  for (int i = 0; i < 4; ++i) {
    int r = wave * 4 + i;
    float* rowp = sE + r * 1024;
    float m = -3.4e38f;
    #pragma unroll
    for (int t = 0; t < 32; ++t)
      m = fmaxf(m, rowp[lane + t * 32] * INV_SQRT_D);
    m = wave_max(m);
    float sum = 0.f;
    #pragma unroll
    for (int t = 0; t < 32; ++t)
      sum += __expf(rowp[lane + t * 32] * INV_SQRT_D - m);
    sum = wave_sum(sum);
    float rinv = 1.0f / sum;
    #pragma unroll
    for (int t = 0; t < 32; ++t) {
      int k = lane + t * 32;
      rowp[k] = __expf(rowp[k] * INV_SQRT_D - m) * rinv;
    }
  }
  __syncthreads();

  // ---- Phase 3: diff = s0 - lam*s1, mask, softmax again -> bf16 ----
  #pragma unroll
  for (int i = 0; i < 2; ++i) {
    int q = wave * 2 + i;
    float dv[32];
    float m = -3.4e38f;
    #pragma unroll
    for (int t = 0; t < 32; ++t) {
      int k = lane + t * 32;
      float d = sE[q * 1024 + k] - lam * sE[(16 + q) * 1024 + k];
      if (k > q0 + q) d = NEGF;
      dv[t] = d;
      m = fmaxf(m, d);
    }
    m = wave_max(m);
    float sum = 0.f;
    #pragma unroll
    for (int t = 0; t < 32; ++t) { dv[t] = __expf(dv[t] - m); sum += dv[t]; }
    sum = wave_sum(sum);
    float rinv = 1.0f / sum;
    #pragma unroll
    for (int t = 0; t < 32; ++t)
      sDiff[q * 1024 + lane + t * 32] = f32_to_bf16(dv[t] * rinv);
  }
  __syncthreads();

  // ---- Phase 4: out(16x128) = diff(16x1024) @ V ; wave w owns dims w*16.. ----
  const int dim = wave * 16 + (lane & 15);
  v8f c = {};
  const u16* vbase = vT + (((size_t)b * CKVH + kvh) * CD + dim) * CS;
  for (int ks = 0; ks < 1024; ks += 32) {
    const u16* ap = sDiff + (lane & 15) * 1024 + ks + off1;
    BF16x16 a;
    a.q[0] = *(const u32x4*)(ap);
    a.q[1] = *(const u32x4*)(ap + 8);
    const u16* bp = vbase + ks + bo16;
    BF16x16 bfr;
    bfr.q[0] = *(const u32x4*)(bp);
    bfr.q[1] = *(const u32x4*)(bp + 8);
    c = wmma_bf16(a, bfr, c);
  }
  #pragma unroll
  for (int j = 0; j < 8; ++j) {
    int q = (lane < 16) ? j : j + 8;
    sOut[q * CD + dim] = c[j];
  }
  __syncthreads();

  // ---- Phase 5: RMSNorm over D, scale by (1-LAM_INIT), store bf16 ----
  #pragma unroll
  for (int i = 0; i < 2; ++i) {
    int q = wave * 2 + i;
    float xv[4], ss = 0.f;
    #pragma unroll
    for (int t = 0; t < 4; ++t) {
      xv[t] = sOut[q * CD + lane * 4 + t];
      ss += xv[t] * xv[t];
    }
    ss = wave_sum(ss);
    float scale = rsqrtf(ss * (1.0f / CD) + 1e-8f) * (1.0f - LAM_INIT);
    #pragma unroll
    for (int t = 0; t < 4; ++t) {
      int d = lane * 4 + t;
      att_ws[((size_t)((((b << 10) + q0 + q) * CH + h))) * CD + d] =
          f32_to_bf16(xv[t] * scale * rms_w[d]);
    }
  }
}

// ---------------------------------------------------------------------------
// Kernel: output projection. out(2048x2048 f32) = att(bf16) @ WoT^T (bf16).
// Same no-LDS dual-accumulator structure as qkv_gemm. grid (128, 16).
// ---------------------------------------------------------------------------
__global__ void out_proj_gemm(const u16* __restrict__ att_ws,
                              const u16* __restrict__ WoT,
                              float* __restrict__ out) {
  const int tid = threadIdx.x, wave = tid >> 5, lane = tid & 31;
  const int m0 = blockIdx.x * 16;
  const int nw = blockIdx.y * 128 + wave * 32;
  const int off1 = (lane < 16) ? 0 : 16;
  const int bo16 = (lane < 16) ? 0 : 16;
  const u16* ap = att_ws + (size_t)(m0 + (lane & 15)) * CE;
  const u16* b0 = WoT + (size_t)(nw + (lane & 15)) * CE;
  const u16* b1 = WoT + (size_t)(nw + 16 + (lane & 15)) * CE;

  v8f c0 = {}, c1 = {};
  for (int k0 = 0; k0 < CE; k0 += 32) {
    if (k0 + 64 < CE) {
      __builtin_prefetch(b0 + k0 + 64, 0, 3);
      __builtin_prefetch(b1 + k0 + 64, 0, 3);
    }
    BF16x16 a, f0, f1;
    a.q[0]  = *(const u32x4*)(ap + k0 + off1);
    a.q[1]  = *(const u32x4*)(ap + k0 + off1 + 8);
    f0.q[0] = *(const u32x4*)(b0 + k0 + bo16);
    f0.q[1] = *(const u32x4*)(b0 + k0 + bo16 + 8);
    f1.q[0] = *(const u32x4*)(b1 + k0 + bo16);
    f1.q[1] = *(const u32x4*)(b1 + k0 + bo16 + 8);
    c0 = wmma_bf16(a, f0, c0);
    c1 = wmma_bf16(a, f1, c1);
  }
  const int n0l = nw + (lane & 15);
  #pragma unroll
  for (int j = 0; j < 8; ++j) {
    int m = m0 + ((lane < 16) ? j : j + 8);
    out[(size_t)m * CE + n0l]      = c0[j];
    out[(size_t)m * CE + n0l + 16] = c1[j];
  }
}

// ---------------------------------------------------------------------------
extern "C" void kernel_launch(void* const* d_in, const int* in_sizes, int n_in,
                              void* d_out, int out_size, void* d_ws, size_t ws_size,
                              hipStream_t stream) {
  (void)in_sizes; (void)n_in; (void)out_size; (void)ws_size;
  const float* x    = (const float*)d_in[0];
  const float* Wq   = (const float*)d_in[1];
  const float* Wk   = (const float*)d_in[2];
  const float* Wv   = (const float*)d_in[3];
  const float* Wo   = (const float*)d_in[4];
  const float* lq1  = (const float*)d_in[5];
  const float* lq2  = (const float*)d_in[6];
  const float* lk1  = (const float*)d_in[7];
  const float* lk2  = (const float*)d_in[8];
  const float* rmsw = (const float*)d_in[9];

  // workspace (bytes):
  //   q_raw 16MB | k_raw 4MB | vT 2MB | att 8MB | lam 64B | xbf 8MB
  //   | WT 26MB (6656x2048 bf16) | WoT 8MB
  char* ws = (char*)d_ws;
  u16*   q_raw = (u16*)(ws);
  u16*   k_raw = (u16*)(ws + 16777216);
  u16*   vT    = (u16*)(ws + 20971520);
  u16*   att   = (u16*)(ws + 23068672);
  float* lam   = (float*)(ws + 31457280);
  u16*   xbf   = (u16*)(ws + 33554432);
  u16*   WT    = (u16*)(ws + 41943040);
  u16*   WoT   = (u16*)(ws + 69206016);

  constexpr int ATTN_SMEM = 32 * 1024 * 4 + 16 * 1024 * 2 + 16 * 128 * 4; // 172032 B
  hipFuncSetAttribute(reinterpret_cast<const void*>(diff_attn_kernel),
                      hipFuncAttributeMaxDynamicSharedMemorySize, ATTN_SMEM);

  // pre-passes: bf16 conversion + weight transposition (whole set is L2-resident)
  cvt_x_bf16<<<8192, 256, 0, stream>>>(x, xbf);
  transpose_w_bf16<<<dim3(64, 128), 256, 0, stream>>>(Wq, NQ,  WT);
  transpose_w_bf16<<<dim3(64,  64), 256, 0, stream>>>(Wk, NKC, WT + (size_t)NQ * CE);
  transpose_w_bf16<<<dim3(64,  16), 256, 0, stream>>>(Wv, NVC, WT + (size_t)(NQ + NKC) * CE);
  transpose_w_bf16<<<dim3(64,  64), 256, 0, stream>>>(Wo, CE,  WoT);
  lambda_kernel<<<1, 32, 0, stream>>>(lq1, lq2, lk1, lk2, lam);

  // main pipeline
  qkv_gemm<<<dim3(128, NTOT / 128), 128, 0, stream>>>(xbf, WT, q_raw, k_raw, vT);
  rope_kernel<<<20480, 256, 0, stream>>>(q_raw, k_raw);
  diff_attn_kernel<<<dim3(64, 16, 2), 256, ATTN_SMEM, stream>>>(q_raw, k_raw, vT,
                                                                lam, rmsw, att);
  out_proj_gemm<<<dim3(128, 16), 128, 0, stream>>>(att, WoT, (float*)d_out);
}
